// RVTSSMStage_40810779247159
// MI455X (gfx1250) — compile-verified, matrix operand order checked by reading, not run on hardware
//
#include <hip/hip_runtime.h>
#include <hip/hip_bf16.h>
#include <math.h>

typedef __bf16 bf16;
typedef __attribute__((ext_vector_type(16))) __bf16 v16bf;
typedef __attribute__((ext_vector_type(8)))  __bf16 bf16x8;
typedef __attribute__((ext_vector_type(4)))  __bf16 bf16x4;
typedef __attribute__((ext_vector_type(8)))  float   v8f;
typedef __attribute__((ext_vector_type(4)))  float   f4;
typedef __attribute__((ext_vector_type(4)))  int     v4i;
typedef __attribute__((address_space(1))) v4i* as1_v4i;
typedef __attribute__((address_space(3))) v4i* as3_v4i;

#define DEV __device__ __forceinline__

#ifndef __has_builtin
#define __has_builtin(x) 0
#endif
#if __has_builtin(__builtin_amdgcn_global_load_async_to_lds_b128)
#define HAVE_ASYNC_LDS 1
#else
#define HAVE_ASYNC_LDS 0
#endif

// ---------------------------------------------------------------------------
// WMMA helpers (CDNA5 16x16x32 bf16, fp32 accumulate)
// ---------------------------------------------------------------------------
DEV v8f wmma_bf16(v16bf a, v16bf b, v8f c) {
  return __builtin_amdgcn_wmma_f32_16x16x32_bf16(false, a, false, b, (short)0, c,
                                                 false, false);
}

// Load one 16x32 bf16 fragment (A or B operand) from a row-major matrix.
// Lane L holds row (L&15); K chunks: e=0..7 -> K=half*8+e, e=8..15 -> 16+half*8+(e-8).
DEV v16bf load_frag(const bf16* base, int ld) {
  int lane = threadIdx.x & 31;
  const bf16* p = base + (size_t)(lane & 15) * ld + ((lane >> 4) << 3);
  bf16x8 c0 = *(const bf16x8*)p;
  bf16x8 c1 = *(const bf16x8*)(p + 16);
  return __builtin_shufflevector(c0, c1, 0, 1, 2, 3, 4, 5, 6, 7,
                                 8, 9, 10, 11, 12, 13, 14, 15);
}

DEV int d_col(int lane) { return lane & 15; }
DEV int d_row(int lane) { return (lane >> 4) << 3; }

DEV float gelu_f(float x) { return 0.5f * x * (1.0f + erff(x * 0.7071067811865475f)); }

// Wave-cooperative LayerNorm of one 128-float row: lane owns cols lane*4..+3.
DEV f4 ln_row(const float* src, int lane, f4 g, f4 bta) {
  f4 v = *(const f4*)(src + lane * 4);
  float s0 = v[0] + v[1] + v[2] + v[3];
#pragma unroll
  for (int off = 16; off; off >>= 1) s0 += __shfl_xor(s0, off);
  float m = s0 * (1.f / 128.f);
  f4 d = v - m;
  float q = d[0] * d[0] + d[1] * d[1] + d[2] * d[2] + d[3] * d[3];
#pragma unroll
  for (int off = 16; off; off >>= 1) q += __shfl_xor(q, off);
  float inv = rsqrtf(q * (1.f / 128.f) + 1e-5f);
  return d * inv * g + bta;
}

// Async (ASYNCcnt-tracked) 16B global->LDS copy with safe fallback.
DEV void copy16_g2l(const float* g, float* l) {
#if HAVE_ASYNC_LDS
  __builtin_amdgcn_global_load_async_to_lds_b128((as1_v4i)g, (as3_v4i)l, 0, 0);
#else
  *(f4*)l = *(const f4*)g;
#endif
}

DEV void async_fence() {
#if HAVE_ASYNC_LDS
#if __has_builtin(__builtin_amdgcn_s_wait_asynccnt)
  __builtin_amdgcn_s_wait_asynccnt(0);
#else
  asm volatile("s_wait_asynccnt 0" ::: "memory");
#endif
#endif
}

// ---------------------------------------------------------------------------
// Small utility kernels
// ---------------------------------------------------------------------------
__global__ void cvt_bf16_kernel(const float* __restrict__ s, bf16* __restrict__ d, int n) {
  int i = blockIdx.x * 256 + threadIdx.x;
  if (i < n) d[i] = (bf16)s[i];
}

// im2col for 2x2/stride-2 conv: one thread per (pixel, input channel);
// writes the 4-tap patch as a single 8-byte bf16x4 store.
__global__ void im2col_kernel(const float* __restrict__ x, bf16* __restrict__ out) {
  size_t i = (size_t)blockIdx.x * 256 + threadIdx.x;
  if (i >= (size_t)147456 * 64) return;
  int c = (int)(i & 63);
  size_t r = i >> 6;
  int n = (int)(r / 2304);
  int rem = (int)(r % 2304);
  int oh = rem / 48, ow = rem % 48;
  const float* p = x + (((size_t)n * 64 + c) * 96 + oh * 2) * 96 + ow * 2;
  float2 t = *(const float2*)p;
  float2 u = *(const float2*)(p + 96);
  bf16x4 o;
  o[0] = (bf16)t.x; o[1] = (bf16)t.y; o[2] = (bf16)u.x; o[3] = (bf16)u.y;
  *(bf16x4*)(out + r * 256 + c * 4) = o;
}

// S5 discretization: lambda_bar = exp(Lam*step), B_bar = ((lb-1)/Lam)*B,
// band-limited C (C_im stored negated so y-GEMM is pure accumulation).
__global__ void s5_prep_kernel(const float* __restrict__ lam_re, const float* __restrict__ lam_im,
                               const float* __restrict__ log_step,
                               const float* __restrict__ B_re, const float* __restrict__ B_im,
                               const float* __restrict__ C_re, const float* __restrict__ C_im,
                               bf16* __restrict__ bbar_re, bf16* __restrict__ bbar_im,
                               bf16* __restrict__ cre_m, bf16* __restrict__ cim_n,
                               float* __restrict__ lb_re_o, float* __restrict__ lb_im_o) {
  int i = blockIdx.x * 256 + threadIdx.x;
  if (i >= 128 * 128) return;
  int p = i >> 7, c = i & 127;
  float step = __expf(log_step[p]);
  float re = lam_re[p], im = lam_im[p];
  float er = __expf(re * step);
  float lbr = er * cosf(im * step), lbi = er * sinf(im * step);
  float den = re * re + im * im;
  float cr = ((lbr - 1.f) * re + lbi * im) / den;
  float ci = (lbi * re - (lbr - 1.f) * im) / den;
  bbar_re[p * 128 + c] = (bf16)(cr * B_re[p * 128 + c] - ci * B_im[p * 128 + c]);
  bbar_im[p * 128 + c] = (bf16)(cr * B_im[p * 128 + c] + ci * B_re[p * 128 + c]);
  float freq = step * 0.15915494309189535f * fabsf(im);
  float mask = (freq < 0.25f) ? 1.f : 0.f;
  cre_m[c * 128 + p] = (bf16)(C_re[c * 128 + p] * mask);
  cim_n[c * 128 + p] = (bf16)(-C_im[c * 128 + p] * mask);
  if (c == 0) { lb_re_o[p] = lbr; lb_im_o[p] = lbi; }
}

// ---------------------------------------------------------------------------
// Conv-as-GEMM (M=147456,K=256,N=128) fused with downsample LayerNorm.
// ---------------------------------------------------------------------------
__global__ __launch_bounds__(256) void conv_gemm_ln_kernel(
    const bf16* __restrict__ im2col, const bf16* __restrict__ wbf,
    const float* __restrict__ bias, const float* __restrict__ lns,
    const float* __restrict__ lnb, float* __restrict__ act) {
  __shared__ __align__(16) float wbuf[64 * 128];  // 32KB
  int r0 = blockIdx.x * 64;
  int tid = threadIdx.x, w = tid >> 5, lane = tid & 31;
  int mi = w >> 1;
#pragma unroll
  for (int j = 0; j < 4; ++j) {
    int ni = 4 * (w & 1) + j;
    v8f acc = {};
#pragma unroll
    for (int kk = 0; kk < 8; ++kk) {
      __builtin_prefetch(wbf + (size_t)(ni * 16) * 256 + kk * 32 + 256, 0, 1);
      v16bf a = load_frag(im2col + (size_t)(r0 + mi * 16) * 256 + kk * 32, 256);
      v16bf b = load_frag(wbf + (size_t)(ni * 16) * 256 + kk * 32, 256);
      acc = wmma_bf16(a, b, acc);
    }
    int col = ni * 16 + d_col(lane), rb = mi * 16 + d_row(lane);
#pragma unroll
    for (int r = 0; r < 8; ++r) wbuf[(rb + r) * 128 + col] = acc[r] + bias[col];
  }
  __syncthreads();
  // wave-cooperative LN: wave w handles rows 8w..8w+7
  f4 g = *(const f4*)(lns + lane * 4), bt = *(const f4*)(lnb + lane * 4);
#pragma unroll
  for (int i = 0; i < 8; ++i) {
    int r = w * 8 + i;
    f4 o = ln_row(wbuf + r * 128, lane, g, bt);
    *(f4*)(act + (size_t)(r0 + r) * 128 + lane * 4) = o;
  }
}

// ---------------------------------------------------------------------------
// Fused MaxViT partition-attention block. One workgroup = one 64-token window.
// ---------------------------------------------------------------------------
__global__ __launch_bounds__(256) void attn_kernel(
    float* __restrict__ act,
    const bf16* __restrict__ qkvw, const float* __restrict__ qkvb,
    const bf16* __restrict__ projw, const float* __restrict__ projb,
    const float* __restrict__ ln1s, const float* __restrict__ ln1b,
    int use_ln1, int grid_mode,
    const float* __restrict__ ln2s, const float* __restrict__ ln2b,
    const bf16* __restrict__ fc1w, const float* __restrict__ fc1b,
    const bf16* __restrict__ fc2w, const float* __restrict__ fc2b) {
  __shared__ __align__(16) char smem[61696];
  bf16* inp = (bf16*)smem;                 // [64][128] LN1(x); ln2 reuse
  bf16* obf = (bf16*)(smem + 16384);       // [64][128] attention out
  bf16* qh = (bf16*)(smem + 32768);        // [64][32]
  bf16* kh = (bf16*)(smem + 36864);        // [64][32]
  bf16* vT = (bf16*)(smem + 40960);        // [32][64]
  float* sc = (float*)(smem + 45056);      // [64][64] scores fp32
  bf16* attnb = (bf16*)(smem + 45056);     // aliases sc
  float* xstage = (float*)(smem + 16384);  // [64][128] fp32 stage (phase 0 only)
  float* wbuf = (float*)(smem + 16384);    // [64][128] fp32 (phase 2)
  bf16* hbf = (bf16*)(smem + 16384);       // [64][256] bf16 (phase 3)
  int* rowidx = (int*)(smem + 61440);      // [64]

  int wid = blockIdx.x;
  int tid = threadIdx.x, w = tid >> 5, lane = tid & 31;

  if (tid < 64) {
    int n = wid / 36, rem = wid % 36, bh = rem / 6, bw = rem % 6;
    int ph = tid >> 3, pw = tid & 7;
    int py, px;
    if (grid_mode) { py = ph * 6 + bh; px = pw * 6 + bw; }
    else           { py = bh * 8 + ph; px = bw * 8 + pw; }
    rowidx[tid] = n * 2304 + py * 48 + px;
  }
  __syncthreads();

  // stage gathered window rows into LDS (async global->LDS when available)
#pragma unroll
  for (int it = 0; it < 8; ++it) {
    int idx = it * 256 + tid;               // 2048 x 16B
    int row = idx >> 5, q4 = (idx & 31) * 4;
    copy16_g2l(act + (size_t)rowidx[row] * 128 + q4, xstage + row * 128 + q4);
  }
  async_fence();
  __syncthreads();

  // inp = LN1(x) (grid) or x (window), bf16; wave w owns rows 8w..8w+7
  {
    f4 g = {}, bt = {};
    if (use_ln1) { g = *(const f4*)(ln1s + lane * 4); bt = *(const f4*)(ln1b + lane * 4); }
#pragma unroll
    for (int i = 0; i < 8; ++i) {
      int r = w * 8 + i;
      f4 o;
      if (use_ln1) o = ln_row(xstage + r * 128, lane, g, bt);
      else         o = *(const f4*)(xstage + r * 128 + lane * 4);
      *(bf16x4*)(inp + r * 128 + lane * 4) = __builtin_convertvector(o, bf16x4);
    }
  }
  __syncthreads();

  // -------- multi-head attention (4 heads, HEAD_DIM=32) --------
  for (int h = 0; h < 4; ++h) {
    int mi = w >> 1, ni = w & 1;
#pragma unroll
    for (int which = 0; which < 3; ++which) {
      int wrow0 = which * 128 + h * 32 + ni * 16;
      v8f acc = {};
#pragma unroll
      for (int kk = 0; kk < 4; ++kk) {
        v16bf a = load_frag(inp + mi * 16 * 128 + kk * 32, 128);
        v16bf b = load_frag(qkvw + (size_t)wrow0 * 128 + kk * 32, 128);
        acc = wmma_bf16(a, b, acc);
      }
      int col = d_col(lane), rb = d_row(lane);
      float bb = qkvb[wrow0 + col];
#pragma unroll
      for (int r = 0; r < 8; ++r) {
        int m = mi * 16 + rb + r;
        int c = ni * 16 + col;
        float val = acc[r] + bb;
        if (which == 0)      qh[m * 32 + c] = (bf16)val;
        else if (which == 1) kh[m * 32 + c] = (bf16)val;
        else                 vT[c * 64 + m] = (bf16)val;  // transposed store
      }
    }
    __syncthreads();

    // scores = (q k^T)/sqrt(32): 16 tiles, 2 per wave, single K=32 step
#pragma unroll
    for (int j = 0; j < 2; ++j) {
      int t8 = 2 * w + j, smi = t8 >> 2, sni = t8 & 3;
      v16bf a = load_frag(qh + smi * 16 * 32, 32);
      v16bf b = load_frag(kh + sni * 16 * 32, 32);
      v8f acc = {};
      acc = wmma_bf16(a, b, acc);
      int col = sni * 16 + d_col(lane), rb = smi * 16 + d_row(lane);
#pragma unroll
      for (int r = 0; r < 8; ++r) sc[(rb + r) * 64 + col] = acc[r] * 0.17677669529663687f;
    }
    __syncthreads();

    // softmax: wave w owns rows 8w..8w+7
    float p0[8], p1[8];
#pragma unroll
    for (int i = 0; i < 8; ++i) {
      int r = 8 * w + i;
      float v0 = sc[r * 64 + lane], v1 = sc[r * 64 + 32 + lane];
      float m = fmaxf(v0, v1);
#pragma unroll
      for (int off = 16; off; off >>= 1) m = fmaxf(m, __shfl_xor(m, off));
      float e0 = __expf(v0 - m), e1 = __expf(v1 - m);
      float s = e0 + e1;
#pragma unroll
      for (int off = 16; off; off >>= 1) s += __shfl_xor(s, off);
      float inv = 1.f / s;
      p0[i] = e0 * inv; p1[i] = e1 * inv;
    }
    __syncthreads();  // all reads of sc done before bf16 alias write
#pragma unroll
    for (int i = 0; i < 8; ++i) {
      int r = 8 * w + i;
      attnb[r * 64 + lane] = (bf16)p0[i];
      attnb[r * 64 + 32 + lane] = (bf16)p1[i];
    }
    __syncthreads();

    // o_h = attn @ v : 8 tiles, 1 per wave, K=64
    {
      int omi = w >> 1, oni = w & 1;
      v8f acc = {};
#pragma unroll
      for (int kk = 0; kk < 2; ++kk) {
        v16bf a = load_frag(attnb + omi * 16 * 64 + kk * 32, 64);
        v16bf b = load_frag(vT + oni * 16 * 64 + kk * 32, 64);
        acc = wmma_bf16(a, b, acc);
      }
      int col = h * 32 + oni * 16 + d_col(lane), rb = omi * 16 + d_row(lane);
#pragma unroll
      for (int r = 0; r < 8; ++r) obf[(rb + r) * 128 + col] = (bf16)acc[r];
    }
    __syncthreads();
  }

  // -------- proj + residual (kept in registers) --------
  v8f wfrag[4];
  int mi = w >> 1;
#pragma unroll
  for (int j = 0; j < 4; ++j) {
    int ni = 4 * (w & 1) + j;
    v8f acc = {};
#pragma unroll
    for (int kk = 0; kk < 4; ++kk) {
      v16bf a = load_frag(obf + mi * 16 * 128 + kk * 32, 128);
      v16bf b = load_frag(projw + (size_t)(ni * 16) * 128 + kk * 32, 128);
      acc = wmma_bf16(a, b, acc);
    }
    int col = ni * 16 + d_col(lane), rb = mi * 16 + d_row(lane);
    float bb = projb[col];
#pragma unroll
    for (int r = 0; r < 8; ++r)
      wfrag[j][r] = acc[r] + bb + act[(size_t)rowidx[rb + r] * 128 + col];
  }
  __syncthreads();  // obf dead; wbuf aliases it
#pragma unroll
  for (int j = 0; j < 4; ++j) {
    int ni = 4 * (w & 1) + j;
    int col = ni * 16 + d_col(lane), rb = mi * 16 + d_row(lane);
#pragma unroll
    for (int r = 0; r < 8; ++r) wbuf[(rb + r) * 128 + col] = wfrag[j][r];
  }
  __syncthreads();

  // LN2 -> inp region (bf16), wave-cooperative
  {
    f4 g = *(const f4*)(ln2s + lane * 4), bt = *(const f4*)(ln2b + lane * 4);
#pragma unroll
    for (int i = 0; i < 8; ++i) {
      int r = w * 8 + i;
      f4 o = ln_row(wbuf + r * 128, lane, g, bt);
      *(bf16x4*)(inp + r * 128 + lane * 4) = __builtin_convertvector(o, bf16x4);
    }
  }
  __syncthreads();

  // -------- MLP: fc1(512)+gelu, fc2, in two 256-wide K halves --------
  v8f facc[4] = {};
  for (int p = 0; p < 2; ++p) {
#pragma unroll
    for (int j = 0; j < 8; ++j) {
      int ni = 8 * (w & 1) + j;
      v8f acc = {};
#pragma unroll
      for (int kk = 0; kk < 4; ++kk) {
        v16bf a = load_frag(inp + mi * 16 * 128 + kk * 32, 128);
        v16bf b = load_frag(fc1w + (size_t)(p * 256 + ni * 16) * 128 + kk * 32, 128);
        acc = wmma_bf16(a, b, acc);
      }
      int col = ni * 16 + d_col(lane), rb = mi * 16 + d_row(lane);
      float bb = fc1b[p * 256 + col];
#pragma unroll
      for (int r = 0; r < 8; ++r) hbf[(rb + r) * 256 + col] = (bf16)gelu_f(acc[r] + bb);
    }
    __syncthreads();
#pragma unroll
    for (int j = 0; j < 4; ++j) {
      int ni = 4 * (w & 1) + j;
#pragma unroll
      for (int kk = 0; kk < 8; ++kk) {
        v16bf a = load_frag(hbf + mi * 16 * 256 + kk * 32, 256);
        v16bf b = load_frag(fc2w + (size_t)(ni * 16) * 512 + p * 256 + kk * 32, 512);
        facc[j] = wmma_bf16(a, b, facc[j]);
      }
    }
    __syncthreads();
  }
#pragma unroll
  for (int j = 0; j < 4; ++j) {
    int ni = 4 * (w & 1) + j;
    int col = ni * 16 + d_col(lane), rb = mi * 16 + d_row(lane);
    float bb = fc2b[col];
#pragma unroll
    for (int r = 0; r < 8; ++r)
      act[(size_t)rowidx[rb + r] * 128 + col] = facc[j][r] + bb + wfrag[j][r];
  }
}

// ---------------------------------------------------------------------------
// Fused S5 block: one workgroup = one sequence (L=16, C=128, P=128).
// ---------------------------------------------------------------------------
__global__ __launch_bounds__(256) void s5_kernel(
    const float* __restrict__ act,
    const float* __restrict__ ln1s, const float* __restrict__ ln1b,
    const bf16* __restrict__ bbar_re, const bf16* __restrict__ bbar_im,
    const float* __restrict__ lb_re, const float* __restrict__ lb_im,
    const bf16* __restrict__ cre_m, const bf16* __restrict__ cim_n,
    const float* __restrict__ Dp,
    const float* __restrict__ ln2s, const float* __restrict__ ln2b,
    const bf16* __restrict__ encw, const float* __restrict__ encb,
    const bf16* __restrict__ decw, const float* __restrict__ decb,
    float* __restrict__ out_y, float* __restrict__ out_str, float* __restrict__ out_sti) {
  __shared__ __align__(16) char sm[45056];
  float* fx  = (float*)sm;              // [16][128] LN1 output (fp32)
  bf16* fxb  = (bf16*)(sm + 8192);      // [16][128]
  float* bur = (float*)(sm + 12288);    // [16][128] Bu real
  float* bui = (float*)(sm + 20480);    // [16][128] Bu imag
  bf16* xsr  = (bf16*)(sm + 28672);     // [16][128] states (bf16)
  bf16* xsi  = (bf16*)(sm + 32768);
  float* xb  = (float*)(sm + 36864);    // [16][128]; also phase-0 stage
  float* fx2 = (float*)(sm + 12288);    // reuse bur
  bf16* fx2b = (bf16*)(sm + 20480);     // reuse bui
  bf16* gb   = (bf16*)(sm + 24576);     // reuse bui upper half

  int s = blockIdx.x;
  int b = s / 2304, yx = s % 2304;
  int tid = threadIdx.x, w = tid >> 5, lane = tid & 31;

  // stage the 16 gathered sequence rows into LDS (async path when available)
#pragma unroll
  for (int it = 0; it < 2; ++it) {
    int idx = it * 256 + tid;               // 512 x 16B
    int row = idx >> 5, q4 = (idx & 31) * 4;
    copy16_g2l(act + (size_t)((row * 4 + b) * 2304 + yx) * 128 + q4,
               xb + row * 128 + q4);
  }
  async_fence();
  __syncthreads();

  // LN1: wave w handles rows 2w, 2w+1
  {
    f4 g = *(const f4*)(ln1s + lane * 4), bt = *(const f4*)(ln1b + lane * 4);
#pragma unroll
    for (int i = 0; i < 2; ++i) {
      int r = w * 2 + i;
      f4 o = ln_row(xb + r * 128, lane, g, bt);
      *(f4*)(fx + r * 128 + lane * 4) = o;
      *(bf16x4*)(fxb + r * 128 + lane * 4) = __builtin_convertvector(o, bf16x4);
    }
  }
  __syncthreads();

  // Bu = fx @ B_bar^T (complex): wave w owns N-tile w for both parts
  {
    v8f ar = {}, ai = {};
#pragma unroll
    for (int kk = 0; kk < 4; ++kk) {
      v16bf a = load_frag(fxb + kk * 32, 128);
      ar = wmma_bf16(a, load_frag(bbar_re + (size_t)(w * 16) * 128 + kk * 32, 128), ar);
      ai = wmma_bf16(a, load_frag(bbar_im + (size_t)(w * 16) * 128 + kk * 32, 128), ai);
    }
    int col = w * 16 + d_col(lane), rb = d_row(lane);
#pragma unroll
    for (int r = 0; r < 8; ++r) {
      bur[(rb + r) * 128 + col] = ar[r];
      bui[(rb + r) * 128 + col] = ai[r];
    }
  }
  __syncthreads();

  // sequential complex scan over L=16, thread p = state index
  if (tid < 128) {
    float sr = 0.f, si = 0.f, ar = lb_re[tid], ai = lb_im[tid];
#pragma unroll
    for (int l = 0; l < 16; ++l) {
      float nr = ar * sr - ai * si + bur[l * 128 + tid];
      float ni = ar * si + ai * sr + bui[l * 128 + tid];
      sr = nr; si = ni;
      xsr[l * 128 + tid] = (bf16)sr;
      xsi[l * 128 + tid] = (bf16)si;
    }
    out_str[((size_t)b * 128 + tid) * 2304 + yx] = sr;
    out_sti[((size_t)b * 128 + tid) * 2304 + yx] = si;
  }
  __syncthreads();

  // y = 2*(xs_re@Cre^T + xs_im@(-Cim)^T) + D*fx ; xb = gelu(y)+fx
  {
    v8f acc = {};
#pragma unroll
    for (int kk = 0; kk < 4; ++kk) {
      acc = wmma_bf16(load_frag(xsr + kk * 32, 128),
                      load_frag(cre_m + (size_t)(w * 16) * 128 + kk * 32, 128), acc);
      acc = wmma_bf16(load_frag(xsi + kk * 32, 128),
                      load_frag(cim_n + (size_t)(w * 16) * 128 + kk * 32, 128), acc);
    }
    int col = w * 16 + d_col(lane), rb = d_row(lane);
#pragma unroll
    for (int r = 0; r < 8; ++r) {
      int l = rb + r;
      float f = fx[l * 128 + col];
      float y = 2.f * acc[r] + Dp[col] * f;
      xb[l * 128 + col] = gelu_f(y) + f;
    }
  }
  __syncthreads();

  // LN2: wave w handles rows 2w, 2w+1
  {
    f4 g = *(const f4*)(ln2s + lane * 4), bt = *(const f4*)(ln2b + lane * 4);
#pragma unroll
    for (int i = 0; i < 2; ++i) {
      int r = w * 2 + i;
      f4 o = ln_row(xb + r * 128, lane, g, bt);
      *(f4*)(fx2 + r * 128 + lane * 4) = o;
      *(bf16x4*)(fx2b + r * 128 + lane * 4) = __builtin_convertvector(o, bf16x4);
    }
  }
  __syncthreads();

  // GEGLU encoder: wave w computes a-tile (cols 16w) and b-tile (cols 128+16w)
  {
    v8f aa = {}, ab = {};
#pragma unroll
    for (int kk = 0; kk < 4; ++kk) {
      v16bf a = load_frag(fx2b + kk * 32, 128);
      aa = wmma_bf16(a, load_frag(encw + (size_t)(w * 16) * 128 + kk * 32, 128), aa);
      ab = wmma_bf16(a, load_frag(encw + (size_t)((w + 8) * 16) * 128 + kk * 32, 128), ab);
    }
    int col = w * 16 + d_col(lane), rb = d_row(lane);
    float ba = encb[col], bb = encb[128 + col];
#pragma unroll
    for (int r = 0; r < 8; ++r) {
      float av = aa[r] + ba, bv = ab[r] + bb;
      gb[(rb + r) * 128 + col] = (bf16)(av * gelu_f(bv));
    }
  }
  __syncthreads();

  // decoder GEMM + residual fx2, channels-first scatter store
  {
    v8f acc = {};
#pragma unroll
    for (int kk = 0; kk < 4; ++kk)
      acc = wmma_bf16(load_frag(gb + kk * 32, 128),
                      load_frag(decw + (size_t)(w * 16) * 128 + kk * 32, 128), acc);
    int col = w * 16 + d_col(lane), rb = d_row(lane);
    float bb = decb[col];
#pragma unroll
    for (int r = 0; r < 8; ++r) {
      int l = rb + r;
      float o = acc[r] + bb + fx2[l * 128 + col];
      out_y[(((size_t)(l * 4 + b)) * 128 + col) * 2304 + yx] = o;
    }
  }
}

// ---------------------------------------------------------------------------
// Host-side orchestration
// ---------------------------------------------------------------------------
enum {
  IN_X = 0, IN_CONV_W, IN_CONV_B, IN_DS_LN_S, IN_DS_LN_B,
  IN_WIN_QKV_W, IN_WIN_QKV_B, IN_WIN_PROJ_W, IN_WIN_PROJ_B,
  IN_WIN_LN2_S, IN_WIN_LN2_B, IN_WIN_FC1_W, IN_WIN_FC1_B,
  IN_WIN_FC2_W, IN_WIN_FC2_B,
  IN_GRID_QKV_W, IN_GRID_QKV_B, IN_GRID_PROJ_W, IN_GRID_PROJ_B,
  IN_GRID_LN2_S, IN_GRID_LN2_B, IN_GRID_FC1_W, IN_GRID_FC1_B,
  IN_GRID_FC2_W, IN_GRID_FC2_B, IN_GRID_LN1_S, IN_GRID_LN1_B,
  IN_S5_LN1_S, IN_S5_LN1_B, IN_S5_LAM_RE, IN_S5_LAM_IM,
  IN_S5_B_RE, IN_S5_B_IM, IN_S5_C_RE, IN_S5_C_IM,
  IN_S5_D, IN_S5_LOG_STEP, IN_S5_LN2_S, IN_S5_LN2_B,
  IN_S5_ENC_W, IN_S5_ENC_B, IN_S5_DEC_W, IN_S5_DEC_B
};

static inline size_t align256(size_t x) { return (x + 255) & ~(size_t)255; }

extern "C" void kernel_launch(void* const* d_in, const int* in_sizes, int n_in,
                              void* d_out, int out_size, void* d_ws, size_t ws_size,
                              hipStream_t stream) {
  (void)in_sizes; (void)n_in; (void)out_size; (void)ws_size;
  char* ws = (char*)d_ws;
  size_t off = 0;
  bf16* im2col = (bf16*)(ws + off);   off = align256(off + (size_t)147456 * 256 * 2);
  float* act   = (float*)(ws + off);  off = align256(off + (size_t)147456 * 128 * 4);

  auto take_bf = [&](size_t elems) { bf16* p = (bf16*)(ws + off); off = align256(off + elems * 2); return p; };
  auto take_f  = [&](size_t elems) { float* p = (float*)(ws + off); off = align256(off + elems * 4); return p; };

  bf16* conv_wb = take_bf(32768);
  bf16* winqkv  = take_bf(49152);
  bf16* winproj = take_bf(16384);
  bf16* winfc1  = take_bf(65536);
  bf16* winfc2  = take_bf(65536);
  bf16* grqkv   = take_bf(49152);
  bf16* grproj  = take_bf(16384);
  bf16* grfc1   = take_bf(65536);
  bf16* grfc2   = take_bf(65536);
  bf16* bbar_re = take_bf(16384);
  bf16* bbar_im = take_bf(16384);
  bf16* cre_m   = take_bf(16384);
  bf16* cim_n   = take_bf(16384);
  bf16* encw    = take_bf(32768);
  bf16* decw    = take_bf(16384);
  float* lbre   = take_f(128);
  float* lbim   = take_f(128);

  struct { const float* s; bf16* d; int n; } cv[] = {
    {(const float*)d_in[IN_CONV_W], conv_wb, 32768},
    {(const float*)d_in[IN_WIN_QKV_W], winqkv, 49152},
    {(const float*)d_in[IN_WIN_PROJ_W], winproj, 16384},
    {(const float*)d_in[IN_WIN_FC1_W], winfc1, 65536},
    {(const float*)d_in[IN_WIN_FC2_W], winfc2, 65536},
    {(const float*)d_in[IN_GRID_QKV_W], grqkv, 49152},
    {(const float*)d_in[IN_GRID_PROJ_W], grproj, 16384},
    {(const float*)d_in[IN_GRID_FC1_W], grfc1, 65536},
    {(const float*)d_in[IN_GRID_FC2_W], grfc2, 65536},
    {(const float*)d_in[IN_S5_ENC_W], encw, 32768},
    {(const float*)d_in[IN_S5_DEC_W], decw, 16384},
  };
  for (auto& c : cv)
    cvt_bf16_kernel<<<(c.n + 255) / 256, 256, 0, stream>>>(c.s, c.d, c.n);

  s5_prep_kernel<<<64, 256, 0, stream>>>(
      (const float*)d_in[IN_S5_LAM_RE], (const float*)d_in[IN_S5_LAM_IM],
      (const float*)d_in[IN_S5_LOG_STEP],
      (const float*)d_in[IN_S5_B_RE], (const float*)d_in[IN_S5_B_IM],
      (const float*)d_in[IN_S5_C_RE], (const float*)d_in[IN_S5_C_IM],
      bbar_re, bbar_im, cre_m, cim_n, lbre, lbim);

  im2col_kernel<<<36864, 256, 0, stream>>>((const float*)d_in[IN_X], im2col);

  conv_gemm_ln_kernel<<<2304, 256, 0, stream>>>(
      im2col, conv_wb, (const float*)d_in[IN_CONV_B],
      (const float*)d_in[IN_DS_LN_S], (const float*)d_in[IN_DS_LN_B], act);

  attn_kernel<<<2304, 256, 0, stream>>>(
      act, winqkv, (const float*)d_in[IN_WIN_QKV_B],
      winproj, (const float*)d_in[IN_WIN_PROJ_B],
      (const float*)d_in[IN_WIN_LN2_S], (const float*)d_in[IN_WIN_LN2_B],
      /*use_ln1=*/0, /*grid_mode=*/0,
      (const float*)d_in[IN_WIN_LN2_S], (const float*)d_in[IN_WIN_LN2_B],
      winfc1, (const float*)d_in[IN_WIN_FC1_B],
      winfc2, (const float*)d_in[IN_WIN_FC2_B]);

  attn_kernel<<<2304, 256, 0, stream>>>(
      act, grqkv, (const float*)d_in[IN_GRID_QKV_B],
      grproj, (const float*)d_in[IN_GRID_PROJ_B],
      (const float*)d_in[IN_GRID_LN1_S], (const float*)d_in[IN_GRID_LN1_B],
      /*use_ln1=*/1, /*grid_mode=*/1,
      (const float*)d_in[IN_GRID_LN2_S], (const float*)d_in[IN_GRID_LN2_B],
      grfc1, (const float*)d_in[IN_GRID_FC1_B],
      grfc2, (const float*)d_in[IN_GRID_FC2_B]);

  float* out = (float*)d_out;
  s5_kernel<<<9216, 256, 0, stream>>>(
      act,
      (const float*)d_in[IN_S5_LN1_S], (const float*)d_in[IN_S5_LN1_B],
      bbar_re, bbar_im, lbre, lbim, cre_m, cim_n,
      (const float*)d_in[IN_S5_D],
      (const float*)d_in[IN_S5_LN2_S], (const float*)d_in[IN_S5_LN2_B],
      encw, (const float*)d_in[IN_S5_ENC_B],
      decw, (const float*)d_in[IN_S5_DEC_B],
      out, out + 18874368, out + 20054016);
}
